// QuantizedLlamaMLP_71167608095143
// MI455X (gfx1250) — compile-verified, matrix operand order checked by reading, not run on hardware
//
#include <hip/hip_runtime.h>

typedef __attribute__((ext_vector_type(16))) __bf16 v16bf;
typedef __attribute__((ext_vector_type(8)))  __bf16 v8bf;
typedef __attribute__((ext_vector_type(8)))  float  v8f;

static constexpr int Hdim = 4096;   // hidden
static constexpr int Idim = 11008;  // intermediate
static constexpr int Mtok = 4 * 2048;
static constexpr int LSTR = 40;     // LDS row stride in bf16 (80B = 16B aligned, bank-conflict-free)

#define SHUF16(lo, hi) __builtin_shufflevector(lo, hi, 0,1,2,3,4,5,6,7,8,9,10,11,12,13,14,15)

__device__ __forceinline__ v16bf load_frag(const __bf16* row, int koff) {
  v8bf lo = *(const v8bf*)(row + koff);        // K = koff .. koff+7   -> VGPR 0-3
  v8bf hi = *(const v8bf*)(row + 16 + koff);   // K = koff+16 .. +23   -> VGPR 4-7
  return SHUF16(lo, hi);
}

__device__ __forceinline__ unsigned pack2(__bf16 a, __bf16 b) {
  union { __bf16 h[2]; unsigned u; } z;
  z.h[0] = a; z.h[1] = b;
  return z.u;
}

// ---------------- Kernel 1: act = silu(x @ Wg) * (x @ Wu), bf16 out ----------------
__global__ __launch_bounds__(256) void mlp_gateup(
    const float* __restrict__ x,
    const int* __restrict__ gate_q, const float* __restrict__ gate_s,
    const int* __restrict__ up_q,   const float* __restrict__ up_s,
    __bf16* __restrict__ act)
{
  __shared__ __align__(16) __bf16 lA[2][128 * LSTR];
  __shared__ __align__(16) __bf16 lG[2][128 * LSTR];
  __shared__ __align__(16) __bf16 lU[2][128 * LSTR];

  const int t    = threadIdx.x;
  const int wave = t >> 5, lane = t & 31;
  const int half = lane >> 4, l16 = lane & 15;
  const int wm = wave & 3;   // wave m-offset = wm*32
  const int wn = wave >> 2;  // wave n-offset = wn*64
  const int m0 = blockIdx.y * 128;
  const int n0 = blockIdx.x * 128;
  const int koff = half * 8;

  v8f accG[2][4], accU[2][4];
  const v8f vzero = {0.f,0.f,0.f,0.f,0.f,0.f,0.f,0.f};
#pragma unroll
  for (int i = 0; i < 2; ++i)
#pragma unroll
    for (int j = 0; j < 4; ++j) { accG[i][j] = vzero; accU[i][j] = vzero; }

  // --- fixed per-thread staging geometry ---
  const int xrow  = t >> 3, xquad = t & 7;   // x: rows xrow + i*32, quad of 4 floats
  const int ncol  = t & 127;                 // weights: one n column per thread
  const int pbase = (t >> 7) * 8;            // 8 consecutive kk-pairs starting here

  // base pointers, advanced by constant strides each K-tile (loop is peeled so no OOB)
  const float* xp  = x      + (size_t)(m0 + xrow) * Hdim + xquad * 4;
  const int*   gqp = gate_q + (size_t)(2 * pbase) * Idim + n0 + ncol;
  const int*   uqp = up_q   + (size_t)(2 * pbase) * Idim + n0 + ncol;
  const float* gsp = gate_s + n0 + ncol;
  const float* usp = up_s   + n0 + ncol;

  float4   xr[4];
  unsigned wgpk[8], wupk[8];  // packed bf16 pairs

  auto stage = [&]() {
#pragma unroll
    for (int i = 0; i < 4; ++i)
      xr[i] = *(const float4*)(xp + (size_t)i * 32 * Hdim);
    const float sg = *gsp, su = *usp;  // one scale load per matrix per tile
#pragma unroll
    for (int i = 0; i < 8; ++i) {
      int q0 = gqp[(size_t)(2 * i)     * Idim];
      int q1 = gqp[(size_t)(2 * i + 1) * Idim];
      wgpk[i] = pack2((__bf16)((float)(q0 - 8) * sg), (__bf16)((float)(q1 - 8) * sg));
      int r0 = uqp[(size_t)(2 * i)     * Idim];
      int r1 = uqp[(size_t)(2 * i + 1) * Idim];
      wupk[i] = pack2((__bf16)((float)(r0 - 8) * su), (__bf16)((float)(r1 - 8) * su));
    }
    xp += 32; gqp += (size_t)32 * Idim; uqp += (size_t)32 * Idim;
  };
  auto store_lds = [&](int buf) {
#pragma unroll
    for (int i = 0; i < 4; ++i) {
      uint2 p;
      p.x = pack2((__bf16)xr[i].x, (__bf16)xr[i].y);
      p.y = pack2((__bf16)xr[i].z, (__bf16)xr[i].w);
      *(uint2*)(&lA[buf][(xrow + i * 32) * LSTR + xquad * 4]) = p;
    }
#pragma unroll
    for (int i = 0; i < 8; ++i) {
      *(unsigned*)(&lG[buf][ncol * LSTR + 2 * (pbase + i)]) = wgpk[i];
      *(unsigned*)(&lU[buf][ncol * LSTR + 2 * (pbase + i)]) = wupk[i];
    }
  };
  auto compute = [&](int buf) {
    v16bf aF[2], gF[4], uF[4];
#pragma unroll
    for (int mt = 0; mt < 2; ++mt)
      aF[mt] = load_frag(&lA[buf][(wm * 32 + mt * 16 + l16) * LSTR], koff);
#pragma unroll
    for (int nt = 0; nt < 4; ++nt) {
      gF[nt] = load_frag(&lG[buf][(wn * 64 + nt * 16 + l16) * LSTR], koff);
      uF[nt] = load_frag(&lU[buf][(wn * 64 + nt * 16 + l16) * LSTR], koff);
    }
#pragma unroll
    for (int mt = 0; mt < 2; ++mt)
#pragma unroll
      for (int nt = 0; nt < 4; ++nt) {
        accG[mt][nt] = __builtin_amdgcn_wmma_f32_16x16x32_bf16(
            false, aF[mt], false, gF[nt], (short)0, accG[mt][nt], false, false);
        accU[mt][nt] = __builtin_amdgcn_wmma_f32_16x16x32_bf16(
            false, aF[mt], false, uF[nt], (short)0, accU[mt][nt], false, false);
      }
  };

  stage();            // tile 0
  store_lds(0);
  __syncthreads();

  int buf = 0;
  for (int ki = 1; ki < Hdim / 32; ++ki) {
    if ((ki & 3) == 0) { gsp += Idim; usp += Idim; }  // next 128-wide scale group
    stage();          // tile ki: global loads overlap the WMMAs below
    compute(buf);     // tile ki-1
    store_lds(buf ^ 1);
    __syncthreads();  // single barrier per K-step
    buf ^= 1;
  }
  compute(buf);       // last tile (peeled: no staging past the end)

  // epilogue: silu(g) * u via fast v_rcp, write bf16 act
#pragma unroll
  for (int mt = 0; mt < 2; ++mt)
#pragma unroll
    for (int nt = 0; nt < 4; ++nt)
#pragma unroll
      for (int i = 0; i < 8; ++i) {
        int m = m0 + wm * 32 + mt * 16 + half * 8 + i;
        int n = n0 + wn * 64 + nt * 16 + l16;
        float g = accG[mt][nt][i];
        float u = accU[mt][nt][i];
        float sig = __builtin_amdgcn_rcpf(1.0f + __expf(-g));
        act[(size_t)m * Idim + n] = (__bf16)(g * sig * u);
      }
}

// ---------------- Kernel 2: out = act @ Wd, f32 out ----------------
__global__ __launch_bounds__(256) void mlp_down(
    const __bf16* __restrict__ act,
    const int* __restrict__ down_q, const float* __restrict__ down_s,
    float* __restrict__ out)
{
  __shared__ __align__(16) __bf16 lA[2][128 * LSTR];
  __shared__ __align__(16) __bf16 lW[2][128 * LSTR];

  const int t    = threadIdx.x;
  const int wave = t >> 5, lane = t & 31;
  const int half = lane >> 4, l16 = lane & 15;
  const int wm = wave & 3;
  const int wn = wave >> 2;
  const int m0 = blockIdx.y * 128;
  const int n0 = blockIdx.x * 128;
  const int koff = half * 8;

  v8f acc[2][4];
  const v8f vzero = {0.f,0.f,0.f,0.f,0.f,0.f,0.f,0.f};
#pragma unroll
  for (int i = 0; i < 2; ++i)
#pragma unroll
    for (int j = 0; j < 4; ++j) acc[i][j] = vzero;

  const int arow  = t >> 2, apart = t & 3;   // act: rows arow + i*64, 8-half chunk
  const int ncol  = t & 127;
  const int pbase = (t >> 7) * 8;

  const __bf16* ap  = act    + (size_t)(m0 + arow) * Idim + apart * 8;
  const int*    dqp = down_q + (size_t)(2 * pbase) * Hdim + n0 + ncol;
  const float*  dsp = down_s + n0 + ncol;

  float4   ar[2];
  unsigned wdpk[8];

  auto stage = [&]() {
#pragma unroll
    for (int i = 0; i < 2; ++i)
      ar[i] = *(const float4*)(ap + (size_t)i * 64 * Idim);
    const float s = *dsp;
#pragma unroll
    for (int i = 0; i < 8; ++i) {
      int q0 = dqp[(size_t)(2 * i)     * Hdim];
      int q1 = dqp[(size_t)(2 * i + 1) * Hdim];
      wdpk[i] = pack2((__bf16)((float)(q0 - 8) * s), (__bf16)((float)(q1 - 8) * s));
    }
    ap += 32; dqp += (size_t)32 * Hdim;
  };
  auto store_lds = [&](int buf) {
#pragma unroll
    for (int i = 0; i < 2; ++i)
      *(float4*)(&lA[buf][(arow + i * 64) * LSTR + apart * 8]) = ar[i];
#pragma unroll
    for (int i = 0; i < 8; ++i)
      *(unsigned*)(&lW[buf][ncol * LSTR + 2 * (pbase + i)]) = wdpk[i];
  };
  auto compute = [&](int buf) {
    v16bf aF[2], wF[4];
#pragma unroll
    for (int mt = 0; mt < 2; ++mt)
      aF[mt] = load_frag(&lA[buf][(wm * 32 + mt * 16 + l16) * LSTR], koff);
#pragma unroll
    for (int nt = 0; nt < 4; ++nt)
      wF[nt] = load_frag(&lW[buf][(wn * 64 + nt * 16 + l16) * LSTR], koff);
#pragma unroll
    for (int mt = 0; mt < 2; ++mt)
#pragma unroll
      for (int nt = 0; nt < 4; ++nt)
        acc[mt][nt] = __builtin_amdgcn_wmma_f32_16x16x32_bf16(
            false, aF[mt], false, wF[nt], (short)0, acc[mt][nt], false, false);
  };

  stage();
  store_lds(0);
  __syncthreads();

  int buf = 0;
  for (int ki = 1; ki < Idim / 32; ++ki) {
    if ((ki & 3) == 0) dsp += Hdim;
    stage();
    compute(buf);
    store_lds(buf ^ 1);
    __syncthreads();
    buf ^= 1;
  }
  compute(buf);

#pragma unroll
  for (int mt = 0; mt < 2; ++mt)
#pragma unroll
    for (int nt = 0; nt < 4; ++nt)
#pragma unroll
      for (int i = 0; i < 8; ++i) {
        int m = m0 + wm * 32 + mt * 16 + half * 8 + i;
        int n = n0 + wn * 64 + nt * 16 + l16;
        out[(size_t)m * Hdim + n] = acc[mt][nt][i];
      }
}

extern "C" void kernel_launch(void* const* d_in, const int* in_sizes, int n_in,
                              void* d_out, int out_size, void* d_ws, size_t ws_size,
                              hipStream_t stream) {
  const float* x      = (const float*)d_in[0];
  const int*   gate_q = (const int*)  d_in[1];
  const float* gate_s = (const float*)d_in[2];
  const int*   up_q   = (const int*)  d_in[3];
  const float* up_s   = (const float*)d_in[4];
  const int*   down_q = (const int*)  d_in[5];
  const float* down_s = (const float*)d_in[6];
  float* out = (float*)d_out;
  __bf16* act = (__bf16*)d_ws;  // requires Mtok*Idim*2 = ~172 MB of workspace

  dim3 blk(256);
  mlp_gateup<<<dim3(Idim / 128, Mtok / 128), blk, 0, stream>>>(
      x, gate_q, gate_s, up_q, up_s, act);
  mlp_down<<<dim3(Hdim / 128, Mtok / 128), blk, 0, stream>>>(
      act, down_q, down_s, out);
}
